// ConditionGuidedCrossAttention_39152921870866
// MI455X (gfx1250) — matrix operand order repaired and checked by reading, not compiled, not run on hardware
//
#include <hip/hip_runtime.h>
#include <hip/hip_bf16.h>

typedef __attribute__((ext_vector_type(16))) _Float16 v16h;
typedef __attribute__((ext_vector_type(8)))  _Float16 v8h;
typedef __attribute__((ext_vector_type(8)))  float    v8f;

// ---------------------------------------------------------------------------
// Problem constants (from reference): B=4, C=512, H=W=32, HEADS=8, GROUPS=32
// seq = 1024, hd = 64, hd_t = 128, keys = 2*seq = 2048
// ---------------------------------------------------------------------------
#define BATCH 4
#define CH    512
#define SEQ   1024
#define NHEAD 8
#define HDV   64
#define HDT   128
#define KEYS  2048

// ---------------------------------------------------------------------------
// f32 -> f16 convert (weights)
// ---------------------------------------------------------------------------
__global__ void f32_to_f16_kernel(const float* __restrict__ in,
                                  _Float16* __restrict__ out, int n) {
    for (int i = blockIdx.x * blockDim.x + threadIdx.x; i < n;
         i += gridDim.x * blockDim.x)
        out[i] = (_Float16)in[i];
}

// pos_emb [512][1024] f32  ->  POS_T [1024 pos][512 ch] f16
__global__ void pos_transpose_kernel(const float* __restrict__ pos,
                                     _Float16* __restrict__ out) {
    for (int i = blockIdx.x * blockDim.x + threadIdx.x; i < CH * SEQ;
         i += gridDim.x * blockDim.x) {
        int c = i / SEQ, p = i % SEQ;
        out[p * CH + c] = (_Float16)pos[i];
    }
}

// ---------------------------------------------------------------------------
// GroupNorm: x [b][512][1024] f32 -> outT [b][1024 pos][512 ch] f16
// one block per (group, batch); 16 channels x 1024 positions per group
// ---------------------------------------------------------------------------
__global__ void groupnorm_kernel(const float* __restrict__ x,
                                 const float* __restrict__ gamma,
                                 const float* __restrict__ beta,
                                 _Float16* __restrict__ outT) {
    const int g = blockIdx.x, b = blockIdx.y;
    const int tid = threadIdx.x;                 // 256 threads
    const int CG = 16;
    const long base = ((long)b * CH + g * CG) * SEQ;
    float s1 = 0.f, s2 = 0.f;
    for (int idx = tid; idx < CG * SEQ; idx += 256) {
        float v = x[base + idx];
        s1 += v; s2 += v * v;
    }
    __shared__ float r1[256], r2[256];
    r1[tid] = s1; r2[tid] = s2;
    __syncthreads();
    for (int off = 128; off > 0; off >>= 1) {
        if (tid < off) { r1[tid] += r1[tid + off]; r2[tid] += r2[tid + off]; }
        __syncthreads();
    }
    const float inv_n = 1.f / (float)(CG * SEQ);
    const float mean = r1[0] * inv_n;
    const float var  = r2[0] * inv_n - mean * mean;
    const float rinv = rsqrtf(var + 1e-6f);
    for (int idx = tid; idx < CG * SEQ; idx += 256) {
        int ch = idx / SEQ, sp = idx % SEQ;
        int c  = g * CG + ch;
        float v = (x[base + idx] - mean) * rinv * gamma[c] + beta[c];
        outT[((long)b * SEQ + sp) * CH + c] = (_Float16)v;
    }
}

// ---------------------------------------------------------------------------
// Generic WMMA GEMM (register-blocked 16M x 64N macro-tile per wave):
//   Out[m][n] = outScale * (sum_k W[m][k] * Xt[n][k] + bias[m])
//   W  : [M][K] f16 row-major (K contiguous)     -> A fragments (loaded once,
//                                                   reused for 4 WMMAs/k-step)
//   Xt : [N][K] f16 row-major (K contiguous)     -> B fragments
// mode 0: f16 outT[(nOff+n)*ldo + mOff + m]      (n-major, contiguous-8 store)
// mode 1: f16 out [(mOff+m)*ldo + nOff + n]      (m-major scatter, for V)
// mode 2: f32 out [m*ldo + n] = acc + bias + residual   (final proj + skip)
// ---------------------------------------------------------------------------
__global__ void wmma_gemm_kernel(const _Float16* __restrict__ W,
                                 const _Float16* __restrict__ Xt,
                                 const float* __restrict__ bias,
                                 float outScale,
                                 _Float16* __restrict__ outH,
                                 float* __restrict__ outF,
                                 const float* __restrict__ residual,
                                 int K, int ldo, int mOff, int nOff, int mode,
                                 long xtStride, long outStride, long resStride) {
    const int b = blockIdx.z;
    const _Float16* xt = Xt + (long)b * xtStride;
    const int L = threadIdx.x, half = L >> 4, l15 = L & 15;
    const int mBase = blockIdx.y * 16, nBase = blockIdx.x * 64;
    const _Float16* arow = W  + (long)(mBase + l15) * K;
    const _Float16* brow = xt + (long)(nBase + l15) * K;
    const long nrs = (long)16 * K;              // row stride between N-tiles

    v8f acc[4] = {{}, {}, {}, {}};
    for (int k0 = 0; k0 < K; k0 += 32) {
        // A fragment: lane row = mBase + l15; chunks [half*16 .. +7], [+8 .. +15]
        v8h a0 = *(const v8h*)(arow + k0 + half * 16);
        v8h a1 = *(const v8h*)(arow + k0 + half * 16 + 8);
        v16h a;
#pragma unroll
        for (int i = 0; i < 8; ++i) { a[i] = a0[i]; a[8 + i] = a1[i]; }
        // 4 B fragments (independent accumulator chains, A reused)
#pragma unroll
        for (int j = 0; j < 4; ++j) {
            v16h bm = *(const v16h*)(brow + j * nrs + k0 + half * 16);
            acc[j] = __builtin_amdgcn_wmma_f32_16x16x32_f16(
                false, a, false, bm, (short)0, acc[j], false, false);
        }
    }

    const int mRow = mBase + half * 8;   // D layout: lane holds rows mRow..mRow+7
    float bsum[8];
#pragma unroll
    for (int i = 0; i < 8; ++i) bsum[i] = bias ? bias[mRow + i] : 0.f;

#pragma unroll
    for (int j = 0; j < 4; ++j) {
        const int n = nBase + j * 16 + l15;
        float r[8];
#pragma unroll
        for (int i = 0; i < 8; ++i) r[i] = (acc[j][i] + bsum[i]) * outScale;

        if (mode == 0) {
            _Float16* dst = outH + (long)b * outStride +
                            (long)(nOff + n) * ldo + mOff + mRow;
#pragma unroll
            for (int i = 0; i < 8; ++i) dst[i] = (_Float16)r[i];
        } else if (mode == 1) {
            _Float16* base = outH + (long)b * outStride;
#pragma unroll
            for (int i = 0; i < 8; ++i)
                base[(long)(mOff + mRow + i) * ldo + nOff + n] = (_Float16)r[i];
        } else {
            float* base = outF + (long)b * outStride;
            const float* res = residual + (long)b * resStride;
#pragma unroll
            for (int i = 0; i < 8; ++i) {
                long idx = (long)(mRow + i) * ldo + n;
                base[idx] = r[i] + res[idx];
            }
        }
    }
}

// ---------------------------------------------------------------------------
// Broadcast (already-scaled) positional projection PT[1024][512] into the
// augmented halves: QT[b][n][512..1023], KT[b][n][512..1023] (both key halves)
// ---------------------------------------------------------------------------
__global__ void pos_broadcast_kernel(const _Float16* __restrict__ PT,
                                     _Float16* __restrict__ QT,
                                     _Float16* __restrict__ KT) {
    const long total = (long)BATCH * KEYS * CH;
    for (long i = blockIdx.x * (long)blockDim.x + threadIdx.x; i < total;
         i += (long)gridDim.x * blockDim.x) {
        int  c = (int)(i % CH);
        long t = i / CH;
        int  n = (int)(t % KEYS);
        int  b = (int)(t / KEYS);
        _Float16 v = PT[(long)(n & (SEQ - 1)) * CH + c];
        KT[((long)b * KEYS + n) * (2 * CH) + CH + c] = v;
        if (n < SEQ)
            QT[((long)b * SEQ + n) * (2 * CH) + CH + c] = v;
    }
}

// ---------------------------------------------------------------------------
// Flash attention: one wave per (batch, head, 16 query rows).
//   QT [b][1024][1024]  (aug channel contiguous; head h => cols h*128..+127)
//   KT [b][2048][1024]
//   Vc [b][512][2048]   (channel-major; head h => rows h*64..+63)
//   outT [b][1024][512] (q-pos-major, channel contiguous -> feeds proj GEMM)
// Online softmax over 2048 keys in 32-key chunks.
// ---------------------------------------------------------------------------
__global__ void attention_kernel(const _Float16* __restrict__ QT,
                                 const _Float16* __restrict__ KT,
                                 const _Float16* __restrict__ Vc,
                                 _Float16* __restrict__ outT) {
    const int qTile = blockIdx.x;              // 64
    const int h     = blockIdx.y;              // 8
    const int b     = blockIdx.z;              // 4
    const int L = threadIdx.x, half = L >> 4, l15 = L & 15;
    const int qBase = qTile * 16;

    // Preload Q A-fragments for the 4 K-steps (hd_t = 128 = 4*32)
    const _Float16* qrow =
        QT + ((long)b * SEQ + qBase + l15) * (2 * CH) + h * HDT;
    v16h aq[4];
#pragma unroll
    for (int kk = 0; kk < 4; ++kk) {
        v8h a0 = *(const v8h*)(qrow + kk * 32 + half * 16);
        v8h a1 = *(const v8h*)(qrow + kk * 32 + half * 16 + 8);
#pragma unroll
        for (int i = 0; i < 8; ++i) { aq[kk][i] = a0[i]; aq[kk][8 + i] = a1[i]; }
    }

    float mrun[8], lrun[8];
    v8f oacc[4];
#pragma unroll
    for (int i = 0; i < 8; ++i) { mrun[i] = -1e30f; lrun[i] = 0.f; }
#pragma unroll
    for (int t = 0; t < 4; ++t) oacc[t] = (v8f){};

    __shared__ _Float16 Plds[16][32];   // P tile staged for A-fragment reload

    const _Float16* kbase = KT + (long)b * KEYS * (2 * CH) + h * HDT;
    const _Float16* vbase = Vc + ((long)b * CH + h * HDV) * KEYS;

    for (int kc = 0; kc < KEYS; kc += 32) {
        // ---- S = (Q*s) (K*s)^T  : two 16-key tiles ----
        v8f st[2];
#pragma unroll
        for (int t = 0; t < 2; ++t) {
            v8f s = {};
            const _Float16* krow = kbase + (long)(kc + t * 16 + l15) * (2 * CH);
#pragma unroll
            for (int kk = 0; kk < 4; ++kk) {
                v16h bm = *(const v16h*)(krow + kk * 32 + half * 16);
                s = __builtin_amdgcn_wmma_f32_16x16x32_f16(
                    false, aq[kk], false, bm, (short)0, s, false, false);
            }
            st[t] = s;
        }

        // ---- online softmax: row stats across the 32 key columns ----
        float mnew[8], alpha[8];
#pragma unroll
        for (int i = 0; i < 8; ++i) {
            float mm = fmaxf(st[0][i], st[1][i]);
#pragma unroll
            for (int off = 1; off < 16; off <<= 1)
                mm = fmaxf(mm, __shfl_xor(mm, off, 32));
            mnew[i]  = fmaxf(mrun[i], mm);
            alpha[i] = __expf(mrun[i] - mnew[i]);
            mrun[i]  = mnew[i];
        }

        __syncthreads();   // previous-iteration Plds reads done
        float rs[8];
#pragma unroll
        for (int i = 0; i < 8; ++i) rs[i] = 0.f;
#pragma unroll
        for (int t = 0; t < 2; ++t)
#pragma unroll
            for (int i = 0; i < 8; ++i) {
                float p = __expf(st[t][i] - mnew[i]);
                rs[i] += p;
                Plds[half * 8 + i][t * 16 + l15] = (_Float16)p;
            }
#pragma unroll
        for (int i = 0; i < 8; ++i) {
            float r = rs[i];
#pragma unroll
            for (int off = 1; off < 16; off <<= 1)
                r += __shfl_xor(r, off, 32);
            lrun[i] = lrun[i] * alpha[i] + r;
        }
#pragma unroll
        for (int t = 0; t < 4; ++t)
#pragma unroll
            for (int i = 0; i < 8; ++i) oacc[t][i] *= alpha[i];
        __syncthreads();   // Plds writes visible

        // ---- reload P as A-fragment (16 q-rows x 32 keys) ----
        v16h ap;
        {
            v8h p0 = *(const v8h*)(&Plds[l15][half * 16]);
            v8h p1 = *(const v8h*)(&Plds[l15][half * 16 + 8]);
#pragma unroll
            for (int i = 0; i < 8; ++i) { ap[i] = p0[i]; ap[8 + i] = p1[i]; }
        }

        // ---- O += P @ V : 4 head-dim tiles of 16 ----
#pragma unroll
        for (int t = 0; t < 4; ++t) {
            const _Float16* vrow =
                vbase + (long)(t * 16 + l15) * KEYS + kc + half * 16;
            v16h bv = *(const v16h*)vrow;
            oacc[t] = __builtin_amdgcn_wmma_f32_16x16x32_f16(
                false, ap, false, bv, (short)0, oacc[t], false, false);
        }
    }

    // ---- finalize and store [b][q][h*64+dv] f16 ----
#pragma unroll
    for (int t = 0; t < 4; ++t)
#pragma unroll
        for (int i = 0; i < 8; ++i) {
            float o = oacc[t][i] / lrun[i];
            outT[((long)b * SEQ + qBase + half * 8 + i) * CH +
                 h * HDV + t * 16 + l15] = (_Float16)o;
        }
}

// ---------------------------------------------------------------------------
// Host launcher
// ---------------------------------------------------------------------------
extern "C" void kernel_launch(void* const* d_in, const int* in_sizes, int n_in,
                              void* d_out, int out_size, void* d_ws,
                              size_t ws_size, hipStream_t stream) {
    const float* x       = (const float*)d_in[0];
    const float* cond    = (const float*)d_in[1];
    const float* norm_g  = (const float*)d_in[2];
    const float* norm_b  = (const float*)d_in[3];
    const float* cnorm_g = (const float*)d_in[4];
    const float* cnorm_b = (const float*)d_in[5];
    const float* qkv_w   = (const float*)d_in[6];
    const float* qkv_b   = (const float*)d_in[7];
    const float* ckv_w   = (const float*)d_in[8];
    const float* ckv_b   = (const float*)d_in[9];
    const float* pos_emb = (const float*)d_in[10];
    const float* pos_w   = (const float*)d_in[11];
    const float* pos_b   = (const float*)d_in[12];
    const float* proj_w  = (const float*)d_in[13];
    const float* proj_b  = (const float*)d_in[14];
    float* out = (float*)d_out;

    // Workspace carve (256-B aligned f16 regions, ~52 MB total)
    char* ws = (char*)d_ws;
    auto alloc16 = [&](long elems) {
        _Float16* p = (_Float16*)ws;
        ws += ((elems * 2 + 255) / 256) * 256;
        return p;
    };
    _Float16* XN    = alloc16((long)BATCH * SEQ * CH);        // [b][s][c]
    _Float16* CN    = alloc16((long)BATCH * SEQ * CH);
    _Float16* POS_T = alloc16((long)SEQ * CH);                // [p][c]
    _Float16* WQKV  = alloc16((long)3 * CH * CH);
    _Float16* WCKV  = alloc16((long)2 * CH * CH);
    _Float16* WPOS  = alloc16((long)CH * CH);
    _Float16* WPRJ  = alloc16((long)CH * CH);
    _Float16* PT    = alloc16((long)SEQ * CH);                // scaled pos proj
    _Float16* QT    = alloc16((long)BATCH * SEQ * 2 * CH);    // [b][s][1024]
    _Float16* KT    = alloc16((long)BATCH * KEYS * 2 * CH);   // [b][2048][1024]
    _Float16* Vc    = alloc16((long)BATCH * CH * KEYS);       // [b][512][2048]
    _Float16* AOT   = alloc16((long)BATCH * SEQ * CH);        // attn out [b][s][c]

    const float scale = powf((float)HDT, -0.25f);  // applied to Q and K each

    // 1) weight converts + pos transpose
    f32_to_f16_kernel<<<3072, 256, 0, stream>>>(qkv_w, WQKV, 3 * CH * CH);
    f32_to_f16_kernel<<<2048, 256, 0, stream>>>(ckv_w, WCKV, 2 * CH * CH);
    f32_to_f16_kernel<<<1024, 256, 0, stream>>>(pos_w, WPOS, CH * CH);
    f32_to_f16_kernel<<<1024, 256, 0, stream>>>(proj_w, WPRJ, CH * CH);
    pos_transpose_kernel<<<2048, 256, 0, stream>>>(pos_emb, POS_T);

    // 2) group norms -> transposed f16 activations
    groupnorm_kernel<<<dim3(32, BATCH), 256, 0, stream>>>(x, norm_g, norm_b, XN);
    groupnorm_kernel<<<dim3(32, BATCH), 256, 0, stream>>>(cond, cnorm_g, cnorm_b, CN);

    // 3) positional projection (batch-independent), pre-scaled
    wmma_gemm_kernel<<<dim3(SEQ / 64, CH / 16, 1), 32, 0, stream>>>(
        WPOS, POS_T, pos_b, scale, PT, nullptr, nullptr,
        CH, CH, 0, 0, /*mode*/0, 0, 0, 0);

    // 4) q / k_x / v_x from normalized x
    wmma_gemm_kernel<<<dim3(SEQ / 64, CH / 16, BATCH), 32, 0, stream>>>(
        WQKV, XN, qkv_b, scale, QT, nullptr, nullptr,
        CH, 2 * CH, 0, 0, /*mode*/0,
        (long)SEQ * CH, (long)SEQ * 2 * CH, 0);
    wmma_gemm_kernel<<<dim3(SEQ / 64, CH / 16, BATCH), 32, 0, stream>>>(
        WQKV + (long)CH * CH, XN, qkv_b + CH, scale, KT, nullptr, nullptr,
        CH, 2 * CH, 0, 0, /*mode*/0,
        (long)SEQ * CH, (long)KEYS * 2 * CH, 0);
    wmma_gemm_kernel<<<dim3(SEQ / 64, CH / 16, BATCH), 32, 0, stream>>>(
        WQKV + (long)2 * CH * CH, XN, qkv_b + 2 * CH, 1.0f, Vc, nullptr, nullptr,
        CH, KEYS, 0, 0, /*mode*/1,
        (long)SEQ * CH, (long)CH * KEYS, 0);

    // 5) k_c / v_c from normalized condition (keys 1024..2047)
    wmma_gemm_kernel<<<dim3(SEQ / 64, CH / 16, BATCH), 32, 0, stream>>>(
        WCKV, CN, ckv_b, scale, KT, nullptr, nullptr,
        CH, 2 * CH, 0, SEQ, /*mode*/0,
        (long)SEQ * CH, (long)KEYS * 2 * CH, 0);
    wmma_gemm_kernel<<<dim3(SEQ / 64, CH / 16, BATCH), 32, 0, stream>>>(
        WCKV + (long)CH * CH, CN, ckv_b + CH, 1.0f, Vc, nullptr, nullptr,
        CH, KEYS, 0, SEQ, /*mode*/1,
        (long)SEQ * CH, (long)CH * KEYS, 0);

    // 6) append scaled positional channels to Q and K
    pos_broadcast_kernel<<<4096, 256, 0, stream>>>(PT, QT, KT);

    // 7) attention (flash, online softmax, all-WMMA)
    attention_kernel<<<dim3(SEQ / 16, NHEAD, BATCH), 32, 0, stream>>>(
        QT, KT, Vc, AOT);

    // 8) output projection + bias + residual, f32 out [b][c][h][w]
    wmma_gemm_kernel<<<dim3(SEQ / 64, CH / 16, BATCH), 32, 0, stream>>>(
        WPRJ, AOT, proj_b, 1.0f, nullptr, out, x,
        CH, SEQ, 0, 0, /*mode*/2,
        (long)SEQ * CH, (long)CH * SEQ, (long)CH * SEQ);
}
